// OlmoLMHeadCE_74380243632868
// MI455X (gfx1250) — matrix operand order repaired and checked by reading, not compiled, not run on hardware
//
#include <hip/hip_runtime.h>
#include <hip/hip_bf16.h>
#include <stdint.h>

// ---------------------------------------------------------------------------
// Fused linear + cross-entropy (sum reduction) for MI455X / gfx1250.
//   x[4096,2048] f32, W[2048,50304] f32, target[4096] i64
//   out[0] = sum((lse - logit[target]) * valid), out[1] = sum(lse * valid)
// bf16 WMMA + TDM double-buffered weight tiles + x resident in LDS,
// online logsumexp in registers. K-tile = 64 (2 WMMA K-steps per barrier).
// ---------------------------------------------------------------------------

typedef __attribute__((ext_vector_type(16))) __bf16       v16bf;
typedef __attribute__((ext_vector_type(8)))  float        v8f;
typedef __attribute__((ext_vector_type(4)))  unsigned int u32x4;
typedef __attribute__((ext_vector_type(8)))  int          i32x8;
typedef __attribute__((ext_vector_type(4)))  int          i32x4;

constexpr int kN = 4096;
constexpr int kD = 2048;
constexpr int kV = 50304;
constexpr int IGNORE_INDEX = -100;

constexpr int ROWS   = 64;                 // rows of x per block
constexpr int XROWB  = 4176;               // padded LDS row stride (bytes) for x (2088 bf16)
constexpr int XB     = ROWS * XROWB;       // 267264 B
constexpr int KTILE  = 64;                 // K elements per TDM tile (2 WMMA steps)
constexpr int BCOLB  = 144;                // LDS stride per weight column: 64*2B + 16B pad
constexpr int VCHUNK = 128;                // columns per V chunk
constexpr int BB     = VCHUNK * BCOLB;     // 18432 B per buffer
constexpr int NKS    = kD / KTILE;         // 32 K-tiles per V chunk
constexpr int NVC    = kV / VCHUNK;        // 393
constexpr int NSTEPS = NVC * NKS;          // 12576
constexpr int MS_OFF = XB + 2 * BB;        // 304128: merge-scratch offset
constexpr int SMEM_BYTES = MS_OFF + (64 * 2 * 3 + 128) * 4;  // 306176 B

union FragBF { int4 q[2]; v16bf v; };
union BF4    { __bf16 h[4]; unsigned long long u; };

// ---------------------------------------------------------------------------
// Kernel 1: transpose + convert W[f32, D x V] -> Wt[bf16, V x D]
// ---------------------------------------------------------------------------
__global__ void olmo_w_transpose_bf16(const float* __restrict__ w,
                                      __bf16* __restrict__ wt) {
  __shared__ __bf16 tile[64][72];  // [v][d] with pad
  const int v0 = blockIdx.x * 64;
  const int d0 = blockIdx.y * 64;
  for (int i = threadIdx.x; i < 64 * 64; i += 256) {
    int d = i >> 6;          // 0..63
    int v = i & 63;          // consecutive threads -> consecutive v (coalesced)
    tile[v][d] = (__bf16)w[(size_t)(d0 + d) * kV + (v0 + v)];
  }
  __syncthreads();
  for (int i = threadIdx.x; i < 64 * 64; i += 256) {
    int v = i >> 6;
    int d = i & 63;          // consecutive threads -> consecutive d (coalesced)
    wt[(size_t)(v0 + v) * kD + (d0 + d)] = tile[v][d];
  }
}

// ---------------------------------------------------------------------------
// TDM: async 2D tile load  Wt[c0..c0+127][kk..kk+63]  ->  LDS (144B col stride)
// ---------------------------------------------------------------------------
__device__ __forceinline__ void tdm_issue(const __bf16* __restrict__ wt,
                                          int step, char* smemBase,
                                          int lane) {
  const int c0 = (step / NKS) * VCHUNK;
  const int kk = (step % NKS) * KTILE;
  char* dst = smemBase + XB + (size_t)(step & 1) * BB;
  const __bf16* src = wt + (size_t)c0 * kD + kk;
#if __has_builtin(__builtin_amdgcn_tensor_load_to_lds)
  (void)lane;
  const unsigned ldsOff = (unsigned)(uintptr_t)dst;
  const unsigned long long ga = (unsigned long long)(uintptr_t)src;
  // D# group 0: count=1 | lds_addr | global_addr | type=2
  u32x4 g0 = { 1u,
               ldsOff,
               (unsigned)(ga & 0xFFFFFFFFu),
               (unsigned)(((ga >> 32) & 0x1FFFFFFu) | 0x80000000u) };
  // D# group 1: data_size=2B, pad_enable, pad_interval=32 DW (code 4),
  //             pad_amount=4 DW (code 3), tensor_dim0=D, tensor_dim1=V,
  //             tile_dim0=64, tile_dim1=128, tensor_dim0_stride=D
  i32x8 g1 = { (int)((1u << 16) | (1u << 20) | (4u << 22) | (3u << 25)),
               (int)(((unsigned)kD & 0xFFFFu) << 16),
               (int)(((unsigned)kD >> 16) | (((unsigned)kV & 0xFFFFu) << 16)),
               (int)(((unsigned)kV >> 16) | ((unsigned)KTILE << 16)),
               (int)VCHUNK,
               (int)kD,
               0, 0 };
  i32x4 gz4 = { 0, 0, 0, 0 };
  i32x8 gz8 = { 0, 0, 0, 0, 0, 0, 0, 0 };
  __builtin_amdgcn_tensor_load_to_lds(g0, g1, gz4, gz4, gz8, 0);
#else
  // Fallback: wave0 copies the 16KB tile with plain b128 loads/stores.
  for (int c = lane; c < VCHUNK; c += 32) {
    const int4* s = (const int4*)(src + (size_t)c * kD);
    int4* d = (int4*)(dst + c * BCOLB);
#pragma unroll
    for (int q = 0; q < 8; ++q) d[q] = s[q];
  }
#endif
}

__device__ __forceinline__ void tdm_wait() {
#if __has_builtin(__builtin_amdgcn_s_wait_tensorcnt)
  __builtin_amdgcn_s_wait_tensorcnt(0);
#endif
}

// ---------------------------------------------------------------------------
// Kernel 2: main fused GEMM + online logsumexp
// grid: kN/ROWS = 64 blocks, 256 threads (8 waves: 4 M-subtiles x 2 V-slices)
// ---------------------------------------------------------------------------
__global__ void olmo_lmhead_ce_main(const float* __restrict__ x,
                                    const __bf16* __restrict__ wt,
                                    const long long* __restrict__ target,
                                    float* __restrict__ partials) {
  extern __shared__ __align__(16) char smem[];
  const int tid    = threadIdx.x;
  const int lane   = tid & 31;
  const int wid    = tid >> 5;
  const int msub   = wid >> 1;      // 0..3  (which 16-row subtile)
  const int vslice = wid & 1;       // 0..1  (which 64-column half of chunk)
  const int rowBlock = blockIdx.x * ROWS;
  const int l16     = lane & 15;
  const int halfHi  = (lane < 16) ? 0 : 1;   // C-layout: VGPR r -> M = r + 8*halfHi

  // ---- stage x rows (fp32 -> bf16) into LDS, padded row stride ----
  for (int i = tid; i < ROWS * kD / 4; i += 256) {
    const int row = i >> 9;                  // / (kD/4)
    const int c4  = (i & 511) << 2;
    const float4 f = *(const float4*)(x + (size_t)(rowBlock + row) * kD + c4);
    BF4 p;
    p.h[0] = (__bf16)f.x; p.h[1] = (__bf16)f.y;
    p.h[2] = (__bf16)f.z; p.h[3] = (__bf16)f.w;
    *(unsigned long long*)(smem + (size_t)row * XROWB + (size_t)c4 * 2) = p.u;
  }

  // ---- per-lane row targets (8 row slots per lane) ----
  int tg[8];
#pragma unroll
  for (int r = 0; r < 8; ++r)
    tg[r] = (int)target[rowBlock + msub * 16 + r + 8 * halfHi];

  float m[8], s[8], tl[8];
#pragma unroll
  for (int r = 0; r < 8; ++r) { m[r] = -3.0e38f; s[r] = 0.f; tl[r] = 0.f; }

  // prologue: issue first weight tile
  if (wid == 0) tdm_issue(wt, 0, smem, lane);
  __syncthreads();  // x staged

  const v8f vzero = { 0.f, 0.f, 0.f, 0.f, 0.f, 0.f, 0.f, 0.f };

  // per-wave invariant LDS bases
  const char* arowBase = smem + (size_t)(msub * 16 + l16) * XROWB +
                         (halfHi ? 16 : 0);
  const int bColOff = (vslice * 64 + l16) * BCOLB + (halfHi ? 16 : 0);

  for (int vc = 0; vc < NVC; ++vc) {
    v8f acc[4];
#pragma unroll
    for (int j = 0; j < 4; ++j) acc[j] = vzero;

    for (int ks = 0; ks < NKS; ++ks) {
      const int step = vc * NKS + ks;
      if (wid == 0) tdm_wait();        // tile[step] resident
      __syncthreads();                 // everyone: buf ready, prev buf free
      if (wid == 0 && step + 1 < NSTEPS) tdm_issue(wt, step + 1, smem, lane);

      const char* bb = smem + XB + (size_t)(step & 1) * BB + bColOff;
      const char* ar = arowBase + (size_t)(ks * KTILE) * 2;

#pragma unroll
      for (int h = 0; h < 2; ++h) {    // two K=32 halves of the K-tile
        // A fragment: 16x32 bf16 (documented VGPR layout)
        FragBF A;
        A.q[0] = *(const int4*)(ar + h * 64);
        A.q[1] = *(const int4*)(ar + h * 64 + 32);
        // load ALL B fragments first so one s_wait_dscnt covers the 4 WMMAs
        FragBF B0, B1, B2, B3;
        B0.q[0] = *(const int4*)(bb + 0 * 16 * BCOLB + h * 64);
        B0.q[1] = *(const int4*)(bb + 0 * 16 * BCOLB + h * 64 + 32);
        B1.q[0] = *(const int4*)(bb + 1 * 16 * BCOLB + h * 64);
        B1.q[1] = *(const int4*)(bb + 1 * 16 * BCOLB + h * 64 + 32);
        B2.q[0] = *(const int4*)(bb + 2 * 16 * BCOLB + h * 64);
        B2.q[1] = *(const int4*)(bb + 2 * 16 * BCOLB + h * 64 + 32);
        B3.q[0] = *(const int4*)(bb + 3 * 16 * BCOLB + h * 64);
        B3.q[1] = *(const int4*)(bb + 3 * 16 * BCOLB + h * 64 + 32);
        acc[0] = __builtin_amdgcn_wmma_f32_16x16x32_bf16(
            false, A.v, false, B0.v, (short)0, acc[0], false, false);
        acc[1] = __builtin_amdgcn_wmma_f32_16x16x32_bf16(
            false, A.v, false, B1.v, (short)0, acc[1], false, false);
        acc[2] = __builtin_amdgcn_wmma_f32_16x16x32_bf16(
            false, A.v, false, B2.v, (short)0, acc[2], false, false);
        acc[3] = __builtin_amdgcn_wmma_f32_16x16x32_bf16(
            false, A.v, false, B3.v, (short)0, acc[3], false, false);
      }
    }

    // ---- online logsumexp update + target-logit capture ----
    const int colBase = vc * VCHUNK + vslice * 64 + l16;
#pragma unroll
    for (int r = 0; r < 8; ++r) {
      const float v0 = acc[0][r], v1 = acc[1][r], v2 = acc[2][r], v3 = acc[3][r];
      const float mx = fmaxf(fmaxf(v0, v1), fmaxf(v2, v3));
      const float nm = fmaxf(m[r], mx);
      s[r] = s[r] * __expf(m[r] - nm) +
             __expf(v0 - nm) + __expf(v1 - nm) +
             __expf(v2 - nm) + __expf(v3 - nm);
      m[r] = nm;
      tl[r] += (colBase +  0 == tg[r]) ? v0 : 0.f;
      tl[r] += (colBase + 16 == tg[r]) ? v1 : 0.f;
      tl[r] += (colBase + 32 == tg[r]) ? v2 : 0.f;
      tl[r] += (colBase + 48 == tg[r]) ? v3 : 0.f;
    }
  }

  // ---- reduce (m,s,tl) across the 16 lanes of each half-wave ----
#pragma unroll
  for (int off = 8; off >= 1; off >>= 1) {
#pragma unroll
    for (int r = 0; r < 8; ++r) {
      const float om = __shfl_xor(m[r], off, 32);
      const float os = __shfl_xor(s[r], off, 32);
      const float ot = __shfl_xor(tl[r], off, 32);
      const float nm = fmaxf(m[r], om);
      s[r]  = s[r] * __expf(m[r] - nm) + os * __expf(om - nm);
      m[r]  = nm;
      tl[r] += ot;
    }
  }

  // ---- merge the two V-slice waves per row via LDS ----
  float* ms_m  = (float*)(smem + MS_OFF);        // [64][2]
  float* ms_s  = ms_m + 128;
  float* ms_tl = ms_s + 128;
  float* red   = ms_tl + 128;                    // [2][64]
  if (l16 == 0) {
#pragma unroll
    for (int r = 0; r < 8; ++r) {
      const int rl = msub * 16 + r + 8 * halfHi;
      ms_m [rl * 2 + vslice] = m[r];
      ms_s [rl * 2 + vslice] = s[r];
      ms_tl[rl * 2 + vslice] = tl[r];
    }
  }
  __syncthreads();
  if (tid < 64) {
    const float m0 = ms_m[tid * 2], m1 = ms_m[tid * 2 + 1];
    const float M  = fmaxf(m0, m1);
    const float S  = ms_s[tid * 2] * __expf(m0 - M) +
                     ms_s[tid * 2 + 1] * __expf(m1 - M);
    const float TL = ms_tl[tid * 2] + ms_tl[tid * 2 + 1];
    const float lse = M + __logf(S);
    const int   t   = (int)target[rowBlock + tid];
    const float valid = (t != IGNORE_INDEX) ? 1.f : 0.f;
    red[tid]      = (lse - TL) * valid;
    red[64 + tid] = lse * valid;
  }
  __syncthreads();
  if (tid == 0) {
    float a = 0.f, b = 0.f;
    for (int i = 0; i < 64; ++i) { a += red[i]; b += red[64 + i]; }
    partials[blockIdx.x * 2 + 0] = a;
    partials[blockIdx.x * 2 + 1] = b;
  }
}

// ---------------------------------------------------------------------------
// Kernel 3: reduce per-block partials -> out[2]
// ---------------------------------------------------------------------------
__global__ void olmo_reduce_partials(const float* __restrict__ p,
                                     float* __restrict__ out) {
  const int t = threadIdx.x;  // 32 threads, 64 blocks of partials
  float a = p[2 * t]     + p[2 * (t + 32)];
  float b = p[2 * t + 1] + p[2 * (t + 32) + 1];
#pragma unroll
  for (int off = 16; off >= 1; off >>= 1) {
    a += __shfl_xor(a, off, 32);
    b += __shfl_xor(b, off, 32);
  }
  if (t == 0) { out[0] = a; out[1] = b; }
}

// ---------------------------------------------------------------------------
extern "C" void kernel_launch(void* const* d_in, const int* in_sizes, int n_in,
                              void* d_out, int out_size, void* d_ws,
                              size_t ws_size, hipStream_t stream) {
  (void)in_sizes; (void)n_in; (void)out_size; (void)ws_size;
  const float*     x   = (const float*)d_in[0];
  const float*     w   = (const float*)d_in[1];
  const long long* tgt = (const long long*)d_in[2];
  float*           out = (float*)d_out;

  char*   ws       = (char*)d_ws;
  __bf16* wt       = (__bf16*)ws;                                // 206,045,184 B
  float*  partials = (float*)(ws + (size_t)kV * (size_t)kD * 2); // 64*2 floats

  // 1) transpose + convert W -> bf16 Wt[V][D]
  dim3 tgrid(kV / 64, kD / 64);
  hipLaunchKernelGGL(olmo_w_transpose_bf16, tgrid, dim3(256), 0, stream, w, wt);

  // 2) fused GEMM + online logsumexp (needs ~300KB dynamic LDS per WGP)
  (void)hipFuncSetAttribute((const void*)olmo_lmhead_ce_main,
                            hipFuncAttributeMaxDynamicSharedMemorySize,
                            SMEM_BYTES);
  hipLaunchKernelGGL(olmo_lmhead_ce_main, dim3(kN / ROWS), dim3(256),
                     SMEM_BYTES, stream, x, wt, tgt, partials);

  // 3) final reduction -> out[0..1]
  hipLaunchKernelGGL(olmo_reduce_partials, dim3(1), dim3(32), 0, stream,
                     partials, out);
}
